// HeatGeneration_8933531976200
// MI455X (gfx1250) — matrix operand order repaired and checked by reading, not compiled, not run on hardware
//
#include <hip/hip_runtime.h>
#include <hip/hip_bf16.h>
#include <math.h>

// ---------------------------------------------------------------------------
// Fused Euler-Maruyama SDE + Stratonovich heat functional for MI455X (gfx1250)
//
// Roofline: ~1.1 TFLOP of 128x128 MLP GEMMs vs ~67 MB of HBM traffic (noise,
// fully L2-resident at 192 MB) -> compute-bound by ~100x. Design:
//  * whole T-loop fused in one kernel (positions never touch HBM),
//  * layer-2 on wave32 WMMA (v_wmma_f32_16x16x32_f16, f32 accumulate); the
//    compiler keeps 16 independent accumulator chains in flight per K-step,
//  * W2 pre-swizzled in LDS to B-fragment order (2x ds_load_b128 / fragment),
//    each fragment loaded once and used by both row tiles,
//  * layer-1 in packed f16 (v_pk_fma_f16 / v_pk_max) straight into A-fragment
//    friendly LDS rows,
//  * layer-3 fused into the WMMA epilogue: bias+ReLU+dot(W3) in registers,
//    reduced with a 16-lane shuffle butterfly (no h2 buffer at all),
//  * sim-MLP(step i) and heat-MLP(step i-1) batched into one 256x128x128 GEMM,
//  * ~104 KB LDS/block -> 3 workgroups per 320 KB WGP (6 waves/SIMD).
// ---------------------------------------------------------------------------

typedef __attribute__((ext_vector_type(16))) _Float16 v16h;
typedef __attribute__((ext_vector_type(8)))  _Float16 v8h;
typedef __attribute__((ext_vector_type(8)))  float    v8f;

union AB { uint4 q[2]; v16h v; };    // two ds_load_b128 chunks -> one WMMA operand
union PH { uint4 q; v8h h; };        // 8 packed halfs <-> one b128

#define Hh     128                   // hidden width
#define SAMPB  128                   // samples per block
#define ROWSM  256                   // GEMM M per step: 128 sim rows + 128 heat rows
#define PITCH  272                   // h1 row pitch bytes (136 halfs), padded vs 256 for LDS banks
#define KF     1.0f                  // spring constant k

// LDS layout (bytes, all 16B aligned)
#define OFF_H1   0
#define OFF_W2F  (OFF_H1  + ROWSM * PITCH)   // 69632 : W2 in B-fragment order (32 KB)
#define OFF_W10  (OFF_W2F + 32768)           // 102400: W1 row 0 as packed f16 [128]
#define OFF_CS   (OFF_W10 + 256)             // c_sim[j]  (f16) = t_sim *W1[1][j]+b1[j]
#define OFF_CH   (OFF_CS  + 256)             // c_heat[j] (f16) = t_heat*W1[1][j]+b1[j]
#define OFF_MID  (OFF_CH  + 256)             // mid, double buffered [2][128] f32
#define OFF_INC  (OFF_MID + 1024)            // inc, double buffered [2][128] f32
#define OFF_YV   (OFF_INC + 1024)            // y per GEMM row [256] f32
#define OFF_RED  (OFF_YV  + 1024)            // block reduction scratch [128] f32
#define LDS_BYTES (OFF_RED + 512)            // 106752 B (~104 KB)

__global__ void __launch_bounds__(256)
heat_sim_kernel(const float* __restrict__ x0, const float* __restrict__ noise,
                const float* __restrict__ W1, const float* __restrict__ b1,
                const float* __restrict__ W2, const float* __restrict__ b2,
                const float* __restrict__ W3, const float* __restrict__ b3,
                float* __restrict__ partial, int N, int T)
{
    extern __shared__ __align__(16) char smem[];
    _Float16* w2f   = (_Float16*)(smem + OFF_W2F);
    _Float16* w10h  = (_Float16*)(smem + OFF_W10);
    _Float16* csimh = (_Float16*)(smem + OFF_CS);
    _Float16* cheath= (_Float16*)(smem + OFF_CH);
    float*    smid  = (float*)(smem + OFF_MID);  // [2][128]
    float*    sinc  = (float*)(smem + OFF_INC);  // [2][128]
    float*    yv    = (float*)(smem + OFF_YV);   // [256]
    float*    red   = (float*)(smem + OFF_RED);

    const int tid  = threadIdx.x;
    const int lane = tid & 31;
    const int wave = tid >> 5;
    const int nc   = lane & 15;
    const int jj   = tid & 127;                  // this thread's j / sample index
    const int gs   = blockIdx.x * SAMPB + jj;

    const int   Tm1 = T - 1;
    const float dt  = 1.0f / (float)Tm1;         // linspace step == all dts
    const float sdt = sqrtf(2.0f * KF * dt);

    // ---- one-time init ----------------------------------------------------
    // W2 -> WMMA B-fragment order: fragment (ct,kc), lane l covers col
    // n = ct*16 + l%16, halfs h=0..15 -> K = kc*32 + (h/8)*16 + ((l&16)?8:0) + h%8,
    // stored as 2 x 16 B chunks so the GEMM loads it with ds_load_b128.
    for (int e = tid; e < Hh * Hh; e += 256) {
        int hc = e & 7, l = (e >> 3) & 31, ch = (e >> 8) & 1;
        int kc = (e >> 9) & 3, ct = e >> 11;
        int K  = kc * 32 + ch * 16 + ((l & 16) ? 8 : 0) + hc;
        w2f[e] = (_Float16)W2[K * Hh + ct * 16 + (l & 15)];
    }
    const float myw11 = W1[Hh + jj];             // register-resident W1 row 1 / b1
    const float myb1  = b1[jj];
    if (tid < 128) {
        w10h[tid] = (_Float16)W1[tid];
        smid[tid] = 0.0f; smid[128 + tid] = 0.0f;
        sinc[tid] = 0.0f; sinc[128 + tid] = 0.0f;
        csimh[tid] = (_Float16)fmaf(dt, myw11, myb1);   // iter 0: t_sim = ts[1] = dt
    } else {
        cheath[jj] = (_Float16)myb1;                    // iter 0: t_heat = 0 (row inactive)
    }
    // per-lane epilogue constants: bias and W3 for the 8 columns this lane owns
    float b2r[8], w3r[8];
    #pragma unroll
    for (int ct = 0; ct < 8; ++ct) { b2r[ct] = b2[ct * 16 + nc]; w3r[ct] = W3[ct * 16 + nc]; }
    const float bb3 = b3[0];
    __syncthreads();

    float x  = (tid < 128) ? x0[gs] : 0.0f;      // sim state (threads 0..127)
    float dh = 0.0f;                             // heat accumulator (threads 128..255)

    // A-fragment base addresses for this wave's two row tiles (wave, wave+8)
    const char* ab0 = smem + OFF_H1 + (wave * 16 + nc) * PITCH + ((lane & 16) ? 16 : 0);
    const char* ab1 = ab0 + 128 * PITCH;

    const v8h zero8 = {(_Float16)0, (_Float16)0, (_Float16)0, (_Float16)0,
                       (_Float16)0, (_Float16)0, (_Float16)0, (_Float16)0};

    // Iteration i: sim step i (valid i < T-1) batched with heat step i-1 (valid i >= 1)
    for (int i = 0; i < T; ++i) {
        const int slot  = i & 1;
        const int pslot = slot ^ 1;

        // ---- layer 1 (K=2) in packed f16: thread tid owns GEMM row tid ----
        {
            const float xin = (tid < 128) ? x : smid[pslot * 128 + jj];
            const _Float16* carr = (tid < 128) ? csimh : cheath;
            const _Float16 xh = (_Float16)xin;
            const v8h xin8 = {xh, xh, xh, xh, xh, xh, xh, xh};
            char* h1row = smem + OFF_H1 + tid * PITCH;
            #pragma unroll
            for (int jb = 0; jb < Hh; jb += 8) {
                PH w, c, r;
                w.q = *(const uint4*)(w10h + jb);
                c.q = *(const uint4*)(carr + jb);
                r.h = __builtin_elementwise_max(
                          __builtin_elementwise_fma(xin8, w.h, c.h), zero8);
                *(uint4*)(h1row + jb * 2) = r.q;   // one ds_store_b128 per 8 halfs
            }
        }
        __syncthreads();

        // ---- layer 2 GEMM (WMMA) with fused layer-3 epilogue --------------
        AB a0[4], a1[4];
        #pragma unroll
        for (int kc = 0; kc < 4; ++kc) {         // A frags for both row tiles
            a0[kc].q[0] = *(const uint4*)(ab0 + kc * 64);
            a0[kc].q[1] = *(const uint4*)(ab0 + kc * 64 + 32);
            a1[kc].q[0] = *(const uint4*)(ab1 + kc * 64);
            a1[kc].q[1] = *(const uint4*)(ab1 + kc * 64 + 32);
        }
        float p0[8] = {0,0,0,0,0,0,0,0};         // per-lane layer-3 partials
        float p1[8] = {0,0,0,0,0,0,0,0};
        #pragma unroll
        for (int ct = 0; ct < 8; ++ct) {
            v8f c0 = {}, c1 = {};
            #pragma unroll
            for (int kc = 0; kc < 4; ++kc) {     // each B fragment loaded ONCE, used twice
                const char* bo = smem + OFF_W2F + (((ct * 4 + kc) * 2) * 32 + lane) * 16;
                AB b;
                b.q[0] = *(const uint4*)(bo);
                b.q[1] = *(const uint4*)(bo + 512);
                c0 = __builtin_amdgcn_wmma_f32_16x16x32_f16(false, a0[kc].v, false, b.v,
                                                            (short)0, c0, false, false);
                c1 = __builtin_amdgcn_wmma_f32_16x16x32_f16(false, a1[kc].v, false, b.v,
                                                            (short)0, c1, false, false);
            }
            #pragma unroll
            for (int r8 = 0; r8 < 8; ++r8) {     // bias + ReLU + dot(W3), in-register
                p0[r8] = fmaf(fmaxf(c0[r8] + b2r[ct], 0.0f), w3r[ct], p0[r8]);
                p1[r8] = fmaf(fmaxf(c1[r8] + b2r[ct], 0.0f), w3r[ct], p1[r8]);
            }
        }
        // 16-lane butterfly: lanes 0..15 -> row rt*16+r, lanes 16..31 -> rt*16+8+r
        #pragma unroll
        for (int r8 = 0; r8 < 8; ++r8) {
            float s0 = p0[r8], s1 = p1[r8];
            #pragma unroll
            for (int off = 1; off < 16; off <<= 1) {
                s0 += __shfl_xor(s0, off, 32);
                s1 += __shfl_xor(s1, off, 32);
            }
            if (nc == r8) {
                const int rsel = (lane & 16) ? 8 : 0;
                yv[wave * 16 + rsel + r8]       = s0;
                yv[(wave + 8) * 16 + rsel + r8] = s1;
            }
        }
        __syncthreads();

        // ---- consume y: heat accumulate / sim update / next-step c --------
        const float y = yv[tid] + bb3;           // mlp output for this thread's row
        if (tid >= 128) {
            if (i >= 1) {
                const float m = smid[pslot * 128 + jj];
                dh = fmaf(sinc[pslot * 128 + jj], -KF * m - 2.0f * KF * y, dh);
            }
            cheath[jj] = (_Float16)fmaf((float)i * dt, myw11, myb1);       // ts[i]
        } else {
            if (i < Tm1) {
                const size_t nidx = (size_t)gs * (size_t)Tm1 + (size_t)i;
                if (i + 16 < Tm1) __builtin_prefetch(noise + nidx + 16, 0, 0);
                const float z  = noise[nidx];
                const float xn = fmaf(KF * x + 2.0f * KF * y, dt, fmaf(z, sdt, x));
                smid[slot * 128 + tid] = 0.5f * (x + xn);
                sinc[slot * 128 + tid] = xn - x;
                x = xn;
            }
            csimh[tid] = (_Float16)fmaf((float)(i + 2) * dt, myw11, myb1); // ts[i+2]
        }
        __syncthreads();
    }

    // ---- deterministic per-block reduction --------------------------------
    if (tid >= 128) red[tid - 128] = dh;
    __syncthreads();
    if (tid == 0) {
        float s = 0.0f;
        for (int k = 0; k < 128; ++k) s += red[k];
        partial[blockIdx.x] = s;
    }
}

__global__ void heat_reduce_kernel(const float* __restrict__ partial,
                                   float* __restrict__ out, int nblocks, int N)
{
    if (threadIdx.x == 0 && blockIdx.x == 0) {
        float s = 0.0f;
        for (int k = 0; k < nblocks; ++k) s += partial[k];   // fixed order: deterministic
        out[0] = s / (float)N;
    }
}

extern "C" void kernel_launch(void* const* d_in, const int* in_sizes, int n_in,
                              void* d_out, int out_size, void* d_ws, size_t ws_size,
                              hipStream_t stream)
{
    const float* x0    = (const float*)d_in[0];
    const float* noise = (const float*)d_in[1];
    const float* W1    = (const float*)d_in[2];
    const float* b1    = (const float*)d_in[3];
    const float* W2    = (const float*)d_in[4];
    const float* b2    = (const float*)d_in[5];
    const float* W3    = (const float*)d_in[6];
    const float* b3    = (const float*)d_in[7];

    const int N = in_sizes[0];                 // 32768
    const int T = in_sizes[1] / N + 1;         // 512
    const int nblocks = N / SAMPB;             // 256

    float* partial = (float*)d_ws;

    (void)hipFuncSetAttribute(reinterpret_cast<const void*>(heat_sim_kernel),
                              hipFuncAttributeMaxDynamicSharedMemorySize,
                              (int)LDS_BYTES);

    heat_sim_kernel<<<nblocks, 256, LDS_BYTES, stream>>>(
        x0, noise, W1, b1, W2, b2, W3, b3, partial, N, T);
    heat_reduce_kernel<<<1, 32, 0, stream>>>(partial, (float*)d_out, nblocks, N);
}
